// WindowAttention_15238543966860
// MI455X (gfx1250) — matrix operand order repaired and checked by reading, not compiled, not run on hardware
//
#include <hip/hip_runtime.h>

// ---------------------------------------------------------------------------
// MI455X (gfx1250) fused window-attention pipeline.
// ~126 GFLOP vs ~0.46 GB HBM floor -> all GEMMs via v_wmma_f32_16x16x32_bf16.
// Weight tiles staged by the Tensor Data Mover (tensor_load_to_lds, double-
// buffered, s_wait_tensorcnt 1 pipelining); bf16 activation tiles via
// global_load_async_to_lds_b128 (s_wait_asynccnt pipelining). Softmax row
// sums computed with WMMA against a constant ones-fragment (no cross-lane
// shuffles -> no ds_bpermute round-trips). fp32 only at the memory edges.
// ---------------------------------------------------------------------------

typedef __attribute__((ext_vector_type(8)))  float        v8f;
typedef __attribute__((ext_vector_type(16))) __bf16       v16bf;
typedef __attribute__((ext_vector_type(4)))  unsigned int u32x4;
typedef __attribute__((ext_vector_type(2)))  unsigned int u32x2;
typedef __attribute__((ext_vector_type(4)))  float        f32x4;
typedef __attribute__((ext_vector_type(8)))  int          i32x8;
typedef __attribute__((ext_vector_type(4)))  int          i32x4;

union Frag { v16bf v; u32x4 q[2]; };

#if __has_builtin(__builtin_amdgcn_tensor_load_to_lds)
#define HAVE_TDM 1
#else
#define HAVE_TDM 0
#endif

__device__ __forceinline__ unsigned short f2bf(float f) {
  unsigned u = __float_as_uint(f);
  u += 0x7FFFu + ((u >> 16) & 1u);           // round-to-nearest-even
  return (unsigned short)(u >> 16);
}

__device__ __forceinline__ v8f wmma_bf16(const Frag& a, const Frag& b, v8f c) {
  return __builtin_amdgcn_wmma_f32_16x16x32_bf16(
      /*neg_a=*/false, a.v, /*neg_b=*/false, b.v,
      /*c_mod=*/(short)0, c, /*reuse_a=*/false, /*reuse_b=*/false);
}

// Async copy global->LDS (ASYNCcnt tracked). LDS offset = low 32 bits of the
// flat shared pointer (LDS aperture maps addr[31:0] -> LDS offset).
__device__ __forceinline__ void async_copy_b128(unsigned lds_byte_off, const void* gaddr) {
  asm volatile("global_load_async_to_lds_b128 %0, %1, off"
               :: "v"(lds_byte_off), "v"(gaddr) : "memory");
}
template <int N>
__device__ __forceinline__ void wait_async_le() {
  asm volatile("s_wait_asynccnt %0" :: "i"(N) : "memory");
}
__device__ __forceinline__ unsigned lds_off_of(const void* p) {
  return (unsigned)(unsigned long long)p;
}

#if HAVE_TDM
// TDM 2D tile load: tile_d1 rows x tile_d0 bf16 elems from a row-major tensor
// (row stride = stride_elems) into LDS. D# pad fields add 16B of LDS padding
// per 64B of data -> destination pitch 80B == ushort[40] rows.
// This toolchain's builtin is the 6-arg form: (g0, g1, g2, g3, g_extra, cpol).
__device__ __forceinline__ void tdm_load_2d(unsigned lds_byte_off, const void* gptr,
                                            unsigned tile_d0, unsigned tile_d1,
                                            unsigned tensor_d0, unsigned tensor_d1,
                                            unsigned stride_elems) {
  unsigned long long ga = (unsigned long long)gptr;
  u32x4 g0;
  g0[0] = 1u;                                           // count=1 (valid), user mode
  g0[1] = lds_byte_off;                                 // lds_addr
  g0[2] = (unsigned)(ga & 0xFFFFFFFFu);                 // global_addr[31:0]
  g0[3] = (unsigned)((ga >> 32) & 0x01FFFFFFu) | (2u << 30);  // addr[56:32] | type=2
  i32x8 g1;
  g1[0] = (int)((1u << 16)        // data_size = 2 bytes
              | (1u << 20)        // pad_enable
              | (3u << 22)        // pad_interval: every 16 DWORDs (64B)
              | (3u << 25));      // pad_amount: 4 DWORDs (16B)
  g1[1] = (int)((tensor_d0 & 0xFFFFu) << 16);           // atomic_barrier=0 | tdim0[15:0]
  g1[2] = (int)(((tensor_d0 >> 16) & 0xFFFFu) | ((tensor_d1 & 0xFFFFu) << 16));
  g1[3] = (int)(((tensor_d1 >> 16) & 0xFFFFu) | ((tile_d0 & 0xFFFFu) << 16));
  g1[4] = (int)(tile_d1 & 0xFFFFu);                     // tile_dim1 | tile_dim2=0
  g1[5] = (int)stride_elems;                            // tensor_dim0_stride[31:0]
  g1[6] = 0;                                            // stride hi | dim1_stride lo
  g1[7] = 0;
  i32x4 gz4 = {0, 0, 0, 0};
  i32x8 gz8 = {0, 0, 0, 0, 0, 0, 0, 0};
  __builtin_amdgcn_tensor_load_to_lds(g0, g1, gz4, gz4, gz8, 0);
}
#endif

// ---------------------------------------------------------------------------
// Weight prep: Wt[n][k] = bf16(W[k][n]); K is always 384 here.
// ---------------------------------------------------------------------------
__global__ void wprep_kernel(const float* __restrict__ W, unsigned short* __restrict__ Wt,
                             int K, int N) {
  int i = blockIdx.x * 256 + threadIdx.x;
  if (i >= K * N) return;
  int n = i / K, k = i - n * K;
  Wt[(size_t)n * K + k] = f2bf(W[(size_t)k * N + n]);
}

// ---------------------------------------------------------------------------
// Fused attention bias: biasp[w16][h][64][64] = rpb_gather + window mask,
// padded with -1e30 so the attention kernel needs no clamps / index math
// (exp(-1e30) == 0 -> pad columns vanish from the softmax row sums).
// ---------------------------------------------------------------------------
__global__ void bias_prep_kernel(const float* __restrict__ mask,   // [16][49][49]
                                 const float* __restrict__ rpb,    // [169][12]
                                 float* __restrict__ biasp) {      // [192][4096]
  int i = blockIdx.x * 256 + threadIdx.x;      // over 192*4096
  int cell = i & 4095, bh = i >> 12;
  int h = bh % 12, w16 = bh / 12;
  int m = cell >> 6, n = cell & 63;
  float v = -1e30f;
  if (m < 49 && n < 49) {
    int mh = m / 7, mw = m - mh * 7;
    int nh = n / 7, nw = n - nh * 7;
    int idx = (mh - nh + 6) * 13 + (mw - nw + 6);
    v = rpb[idx * 12 + h] + mask[w16 * 2401 + m * 49 + n];
  }
  biasp[i] = v;
}

// ---------------------------------------------------------------------------
// Generic WMMA GEMM: C[M,N] = A[M,K] @ W[K,N] + bias.
//   A: fp32 (AF32=1, converted to bf16 in LDS) or bf16 (AF32=0, async-to-LDS)
//   Wt: pre-transposed bf16 [N][K], tiles staged by TDM
//   C: bf16 (OF32=0) or fp32 (OF32=1)
// Block tile 128x128, 8 waves; wave w: rows w*16..+15 x 8 N-frags; K step 32.
// Double-buffered LDS: stage(i+1) overlaps compute(i); the in-order TENSORcnt
// / ASYNCcnt semantics let us wait "<= one stage in flight".
// ---------------------------------------------------------------------------
template <int AF32, int OF32>
__global__ __launch_bounds__(256) void gemm_bf16_kernel(
    const void* __restrict__ Aptr, const unsigned short* __restrict__ Wt,
    const float* __restrict__ bias, void* __restrict__ Cptr,
    int M, int N, int K) {
  __shared__ unsigned short As[2][128][40];   // 32 K + 8 pad (80B pitch)
  __shared__ unsigned short Bs[2][128][40];

  const int t    = threadIdx.x;
  const int lane = t & 31;
  const int w    = t >> 5;
  const int hi   = lane >> 4;
  const int l15  = lane & 15;
  const int r0   = blockIdx.x * 128;
  const int n0   = blockIdx.y * 128;
  const int KSTEPS = K >> 5;

  // async-to-LDS ops issued per thread per stage (for the <=N pipeline wait)
  constexpr int APS = (AF32 ? 0 : 2) + (HAVE_TDM ? 0 : 2);

  auto stageB = [&](int step, int buf) {
#if HAVE_TDM
    if (w == 0)
      tdm_load_2d(lds_off_of(&Bs[buf][0][0]), Wt + (size_t)n0 * K + step * 32,
                  /*tile_d0=*/32, /*tile_d1=*/128,
                  /*tensor_d0=*/(unsigned)K, /*tensor_d1=*/(unsigned)N,
                  /*stride=*/(unsigned)K);
#else
    #pragma unroll
    for (int j = 0; j < 2; ++j) {
      int idx = t * 2 + j;
      int row = idx >> 2, ch = idx & 3;
      async_copy_b128(lds_off_of(&Bs[buf][row][ch * 8]),
                      Wt + (size_t)(n0 + row) * K + step * 32 + ch * 8);
    }
#endif
  };
  auto stageA = [&](int step, int buf) {
    if (AF32) {
      const float* A = (const float*)Aptr;
      #pragma unroll
      for (int j = 0; j < 4; ++j) {
        int lin = t + j * 256;
        int row = lin >> 3, c4 = (lin & 7) * 4;
        f32x4 f = *(const f32x4*)(A + (size_t)(r0 + row) * K + step * 32 + c4);
        unsigned p0 = (unsigned)f2bf(f.x) | ((unsigned)f2bf(f.y) << 16);
        unsigned p1 = (unsigned)f2bf(f.z) | ((unsigned)f2bf(f.w) << 16);
        u32x2 pk = { p0, p1 };
        *(u32x2*)&As[buf][row][c4] = pk;
      }
    } else {
      const unsigned short* A = (const unsigned short*)Aptr;
      #pragma unroll
      for (int j = 0; j < 2; ++j) {
        int idx = t * 2 + j;
        int row = idx >> 2, ch = idx & 3;
        async_copy_b128(lds_off_of(&As[buf][row][ch * 8]),
                        A + (size_t)(r0 + row) * K + step * 32 + ch * 8);
      }
    }
  };

  v8f acc[8] = {};

  stageB(0, 0);
  stageA(0, 0);
  for (int i = 0; i < KSTEPS; ++i) {
    const int cur = i & 1;
    const bool more = (i + 1) < KSTEPS;
    if (more) {                       // overlap next stage with this compute
      stageB(i + 1, cur ^ 1);
      stageA(i + 1, cur ^ 1);
    }
    // in-order counters: "<= one stage outstanding" means stage i is resident
    if (APS) { if (more) wait_async_le<APS>(); else wait_async_le<0>(); }
#if HAVE_TDM
    if (more) __builtin_amdgcn_s_wait_tensorcnt(1);
    else      __builtin_amdgcn_s_wait_tensorcnt(0);
#endif
    __syncthreads();

    // --- compute: one A frag + 8 B frags -> 8 WMMAs ------------------------
    Frag af;
    const int am = w * 16 + l15;
    af.q[0] = *(const u32x4*)&As[cur][am][hi ? 8 : 0];
    af.q[1] = *(const u32x4*)&As[cur][am][hi ? 24 : 16];
    #pragma unroll
    for (int nt = 0; nt < 8; ++nt) {
      Frag bf_;
      const int bn = nt * 16 + l15;
      bf_.q[0] = *(const u32x4*)&Bs[cur][bn][hi ? 8 : 0];
      bf_.q[1] = *(const u32x4*)&Bs[cur][bn][hi ? 24 : 16];
      acc[nt] = wmma_bf16(af, bf_, acc[nt]);
    }
    __syncthreads();
  }

  // --- epilogue: C layout = rows in VGPRs (r / r+8), column per lane -------
  #pragma unroll
  for (int nt = 0; nt < 8; ++nt) {
    const int col = n0 + nt * 16 + l15;
    const float bv = bias[col];
    #pragma unroll
    for (int r = 0; r < 8; ++r) {
      const int row = r0 + w * 16 + r + hi * 8;
      const float val = acc[nt][r] + bv;
      if (OF32) ((float*)Cptr)[(size_t)row * N + col] = val;
      else      ((unsigned short*)Cptr)[(size_t)row * N + col] = f2bf(val);
    }
  }
}

// ---------------------------------------------------------------------------
// Attention: 1 block per window, 4 waves (grid.y=3 -> 12 heads).
// Per wave/head: S = q k^T (16 WMMAs, 64x64 padded) + fused padded bias,
// e = exp(S) -> LDS bf16; softmax row sums via WMMA against a constant
// ones-fragment (row sums land in the same C-layout rows, so 1/sum stays in
// registers and is applied as a row scale on O). O = P~ @ V for 3 value
// streams (16 WMMAs each). No cross-lane shuffles anywhere.
// ---------------------------------------------------------------------------
__global__ __launch_bounds__(128) void attn_kernel(
    const unsigned short* __restrict__ qkv,   // [M][1152] bf16
    const unsigned short* __restrict__ vsc,   // [M][384]  bf16
    const unsigned short* __restrict__ vsh,   // [M][384]  bf16
    const float* __restrict__ biasp,          // [16][12][64][64], pad = -1e30
    unsigned short* __restrict__ ax,
    unsigned short* __restrict__ asc,
    unsigned short* __restrict__ ash) {
  __shared__ unsigned short Pw[4][64][72];
  __shared__ unsigned short Vt[4][32][72];

  const int w    = threadIdx.x >> 5;
  const int lane = threadIdx.x & 31;
  const int hi   = lane >> 4;
  const int l15  = lane & 15;
  const int b    = blockIdx.x;
  const int h    = blockIdx.y * 4 + w;
  const float SC = 0.17677669529663687f;     // 32^-0.5

  const unsigned short* qb = qkv + (size_t)b * 49 * 1152 + h * 32;
  const float* bp = biasp + ((size_t)(b & 15) * 12 + h) * 4096;

  // constant all-ones bf16 B fragment (1.0bf16 = 0x3F80)
  Frag ones;
  ones.q[0] = (u32x4){0x3F803F80u, 0x3F803F80u, 0x3F803F80u, 0x3F803F80u};
  ones.q[1] = ones.q[0];

  // k fragments are mt-invariant: load once.
  Frag kf[4];
  #pragma unroll
  for (int nt = 0; nt < 4; ++nt) {
    const int ne = nt * 16 + l15;
    const int tk = ne > 48 ? 48 : ne;
    const unsigned short* kr = qb + 384 + (size_t)tk * 1152 + (hi ? 8 : 0);
    kf[nt].q[0] = *(const u32x4*)(kr);
    kf[nt].q[1] = *(const u32x4*)(kr + 16);
  }

  v8f invs[4];                               // 1/rowsum per (mt, r)

  // ---------------- phase 1: P~ = exp(q k^T * SC + bias) -------------------
  for (int mt = 0; mt < 4; ++mt) {
    const int me = mt * 16 + l15;
    const int tq = me > 48 ? 48 : me;
    const unsigned short* qr = qb + (size_t)tq * 1152;
    Frag qf;
    qf.q[0] = *(const u32x4*)(qr + (hi ? 8 : 0));
    qf.q[1] = *(const u32x4*)(qr + (hi ? 24 : 16));

    v8f s[4];
    #pragma unroll
    for (int nt = 0; nt < 4; ++nt) {
      v8f z = {};
      s[nt] = wmma_bf16(qf, kf[nt], z);
    }

    // fused scale + (rpb + mask + pad) bias, then exp; store P~ to LDS
    #pragma unroll
    for (int nt = 0; nt < 4; ++nt) {
      const float* bcol = bp + nt * 16 + l15;
      #pragma unroll
      for (int r = 0; r < 8; ++r) {
        const int m_e = mt * 16 + r + hi * 8;
        s[nt][r] = __expf(s[nt][r] * SC + bcol[m_e * 64]);
      }
    }
    #pragma unroll
    for (int nt = 0; nt < 4; ++nt)
      #pragma unroll
      for (int r = 0; r < 8; ++r)
        Pw[w][mt * 16 + r + hi * 8][nt * 16 + l15] = f2bf(s[nt][r]);

    // row sums via WMMA: P~(16x64) @ ones(64x16); every output column holds
    // the row sum, in the same r/lane row mapping as the O fragments.
    const int mr = mt * 16 + l15;
    Frag pa0, pa1;
    pa0.q[0] = *(const u32x4*)&Pw[w][mr][hi ? 8 : 0];
    pa0.q[1] = *(const u32x4*)&Pw[w][mr][hi ? 24 : 16];
    pa1.q[0] = *(const u32x4*)&Pw[w][mr][32 + (hi ? 8 : 0)];
    pa1.q[1] = *(const u32x4*)&Pw[w][mr][32 + (hi ? 24 : 16)];
    v8f rs = {};
    rs = wmma_bf16(pa0, ones, rs);
    rs = wmma_bf16(pa1, ones, rs);
    #pragma unroll
    for (int r = 0; r < 8; ++r) invs[mt][r] = 1.f / rs[r];
  }

  // ---------------- phase 2: O = (P~ @ V) * inv_rowsum ---------------------
  for (int st = 0; st < 3; ++st) {
    const unsigned short* vb;
    int vstr;
    if (st == 0)      { vb = qkv + (size_t)b * 49 * 1152 + 768 + h * 32; vstr = 1152; }
    else if (st == 1) { vb = vsc + (size_t)b * 49 * 384 + h * 32;        vstr = 384; }
    else              { vb = vsh + (size_t)b * 49 * 384 + h * 32;        vstr = 384; }
    unsigned short* ob = (st == 0) ? ax : (st == 1) ? asc : ash;

    // stage V^T: Vt[d=lane][tok], pad tokens 49..63 with zeros
    for (int tok = 0; tok < 64; ++tok) {
      unsigned short val = (tok < 49) ? vb[(size_t)tok * vstr + lane] : (unsigned short)0;
      Vt[w][lane][tok] = val;
    }

    for (int mt = 0; mt < 4; ++mt) {
      const int mr = mt * 16 + l15;
      Frag p0, p1;
      p0.q[0] = *(const u32x4*)&Pw[w][mr][hi ? 8 : 0];
      p0.q[1] = *(const u32x4*)&Pw[w][mr][hi ? 24 : 16];
      p1.q[0] = *(const u32x4*)&Pw[w][mr][32 + (hi ? 8 : 0)];
      p1.q[1] = *(const u32x4*)&Pw[w][mr][32 + (hi ? 24 : 16)];
      #pragma unroll
      for (int nv = 0; nv < 2; ++nv) {
        const int dcol = nv * 16 + l15;
        Frag b0, b1;
        b0.q[0] = *(const u32x4*)&Vt[w][dcol][hi ? 8 : 0];
        b0.q[1] = *(const u32x4*)&Vt[w][dcol][hi ? 24 : 16];
        b1.q[0] = *(const u32x4*)&Vt[w][dcol][32 + (hi ? 8 : 0)];
        b1.q[1] = *(const u32x4*)&Vt[w][dcol][32 + (hi ? 24 : 16)];
        v8f o = {};
        o = wmma_bf16(p0, b0, o);
        o = wmma_bf16(p1, b1, o);
        #pragma unroll
        for (int r = 0; r < 8; ++r) {
          const int tg = mt * 16 + r + hi * 8;
          if (tg < 49)
            ob[((size_t)b * 49 + tg) * 384 + h * 32 + dcol] = f2bf(o[r] * invs[mt][r]);
        }
      }
    }
  }
}

// ---------------------------------------------------------------------------
extern "C" void kernel_launch(void* const* d_in, const int* in_sizes, int n_in,
                              void* d_out, int out_size, void* d_ws, size_t ws_size,
                              hipStream_t stream) {
  (void)in_sizes; (void)n_in; (void)out_size; (void)ws_size;
  const float* x     = (const float*)d_in[0];
  const float* scale = (const float*)d_in[1];
  const float* shift = (const float*)d_in[2];
  const float* mask  = (const float*)d_in[3];
  const float* qkv_w = (const float*)d_in[4];
  const float* qkv_b = (const float*)d_in[5];
  const float* vsc_w = (const float*)d_in[6];
  const float* vsc_b = (const float*)d_in[7];
  const float* vsh_w = (const float*)d_in[8];
  const float* vsh_b = (const float*)d_in[9];
  const float* rpb   = (const float*)d_in[10];
  const float* px_w  = (const float*)d_in[11];
  const float* px_b  = (const float*)d_in[12];
  const float* ps_w  = (const float*)d_in[13];
  const float* ps_b  = (const float*)d_in[14];
  const float* ph_w  = (const float*)d_in[15];
  const float* ph_b  = (const float*)d_in[16];

  const int M = 1024 * 49;                   // 50176 rows
  char* ws = (char*)d_ws;
  size_t cursor = 0;
  auto alloc = [&](size_t bytes) -> void* {
    void* p = ws + cursor;
    cursor = (cursor + bytes + 255) & ~(size_t)255;
    return p;
  };

  unsigned short* qkvb = (unsigned short*)alloc((size_t)M * 1152 * 2);
  unsigned short* vscb = (unsigned short*)alloc((size_t)M * 384 * 2);
  unsigned short* vshb = (unsigned short*)alloc((size_t)M * 384 * 2);
  unsigned short* axb  = (unsigned short*)alloc((size_t)M * 384 * 2);
  unsigned short* ascb = (unsigned short*)alloc((size_t)M * 384 * 2);
  unsigned short* ashb = (unsigned short*)alloc((size_t)M * 384 * 2);
  unsigned short* wt_qkv = (unsigned short*)alloc((size_t)384 * 1152 * 2);
  unsigned short* wt_vsc = (unsigned short*)alloc((size_t)384 * 384 * 2);
  unsigned short* wt_vsh = (unsigned short*)alloc((size_t)384 * 384 * 2);
  unsigned short* wt_px  = (unsigned short*)alloc((size_t)384 * 384 * 2);
  unsigned short* wt_ps  = (unsigned short*)alloc((size_t)384 * 384 * 2);
  unsigned short* wt_ph  = (unsigned short*)alloc((size_t)384 * 384 * 2);
  float*          biasp  = (float*)alloc((size_t)192 * 4096 * 4);

  // weight transpose + bf16 convert, fused bias table
  wprep_kernel<<<(384 * 1152 + 255) / 256, 256, 0, stream>>>(qkv_w, wt_qkv, 384, 1152);
  wprep_kernel<<<(384 * 384 + 255) / 256, 256, 0, stream>>>(vsc_w, wt_vsc, 384, 384);
  wprep_kernel<<<(384 * 384 + 255) / 256, 256, 0, stream>>>(vsh_w, wt_vsh, 384, 384);
  wprep_kernel<<<(384 * 384 + 255) / 256, 256, 0, stream>>>(px_w, wt_px, 384, 384);
  wprep_kernel<<<(384 * 384 + 255) / 256, 256, 0, stream>>>(ps_w, wt_ps, 384, 384);
  wprep_kernel<<<(384 * 384 + 255) / 256, 256, 0, stream>>>(ph_w, wt_ph, 384, 384);
  bias_prep_kernel<<<(192 * 4096) / 256, 256, 0, stream>>>(mask, rpb, biasp);

  const dim3 blk(256);
  // input projections (fp32 A -> bf16 out)
  gemm_bf16_kernel<1, 0><<<dim3(M / 128, 1152 / 128), blk, 0, stream>>>(
      x, wt_qkv, qkv_b, qkvb, M, 1152, 384);
  gemm_bf16_kernel<1, 0><<<dim3(M / 128, 3), blk, 0, stream>>>(
      scale, wt_vsc, vsc_b, vscb, M, 384, 384);
  gemm_bf16_kernel<1, 0><<<dim3(M / 128, 3), blk, 0, stream>>>(
      shift, wt_vsh, vsh_b, vshb, M, 384, 384);

  // attention core
  attn_kernel<<<dim3(1024, 3), dim3(128), 0, stream>>>(
      qkvb, vscb, vshb, biasp, axb, ascb, ashb);

  // output projections (bf16 A -> fp32 out into d_out, concatenated)
  float* out = (float*)d_out;
  const size_t OS = (size_t)M * 384;
  gemm_bf16_kernel<0, 1><<<dim3(M / 128, 3), blk, 0, stream>>>(
      axb, wt_px, px_b, out, M, 384, 384);
  gemm_bf16_kernel<0, 1><<<dim3(M / 128, 3), blk, 0, stream>>>(
      ascb, wt_ps, ps_b, out + OS, M, 384, 384);
  gemm_bf16_kernel<0, 1><<<dim3(M / 128, 3), blk, 0, stream>>>(
      ashb, wt_ph, ph_b, out + 2 * OS, M, 384, 384);
}
